// LSTM_model_89215060672788
// MI455X (gfx1250) — matrix operand order repaired and checked by reading, not compiled, not run on hardware
//
#include <hip/hip_runtime.h>

// ---- problem dims ----
#define IN   128   // I
#define HD   512   // H
#define TT   256   // T
#define BT   512   // B (batch)

typedef __attribute__((ext_vector_type(16))) __bf16 v16bf;
typedef __attribute__((ext_vector_type(8)))  __bf16 v8bf;
typedef __attribute__((ext_vector_type(8)))  float  v8f;

union AFrag {
    v16bf v;
    v8bf  h8[2];
};

// Hardware tanh (v_tanh_f32 on gfx1250) with compile-safe fallbacks.
__device__ __forceinline__ float htanh_(float x) {
#if __has_builtin(__builtin_amdgcn_tanhf)
    return __builtin_amdgcn_tanhf(x);
#elif __has_builtin(__builtin_amdgcn_tanh_f32)
    return __builtin_amdgcn_tanh_f32(x);
#else
    return tanhf(x);
#endif
}
// sigmoid(x) = 0.5 * tanh(x/2) + 0.5  -> 1 TRANS + mul + fma
__device__ __forceinline__ float sigmoidf_(float x) {
    return __builtin_fmaf(0.5f, htanh_(0.5f * x), 0.5f);
}

// ---------------------------------------------------------------------------
// one-time helpers
// ---------------------------------------------------------------------------
__global__ void cvt_f32_to_bf16(const float* __restrict__ s, __bf16* __restrict__ d, int n) {
    int i = blockIdx.x * blockDim.x + threadIdx.x;
    if (i < n) d[i] = (__bf16)s[i];
}
__global__ void zero_bf16_k(__bf16* __restrict__ p, int n) {
    int i = blockIdx.x * blockDim.x + threadIdx.x;
    if (i < n) p[i] = (__bf16)0.0f;
}
__global__ void zero_f32_k(float* __restrict__ p, int n) {
    int i = blockIdx.x * blockDim.x + threadIdx.x;
    if (i < n) p[i] = 0.0f;
}

// ---------------------------------------------------------------------------
// Kernel 1: gates = x_t @ W_ih^T + h_prev @ W_hh^T + b_ih + b_hh,
// then LSTM cell update. One workgroup = 64 batch rows x 64 H-columns,
// covering the matching 64-col slice of all 4 gates (i,f,g,o).
// 256 threads = 8 waves, arranged 4 (row tiles) x 2 (col tiles of 32).
// ---------------------------------------------------------------------------
__global__ __launch_bounds__(256)
void lstm_gates_cell(const float* __restrict__ x, int t,
                     const __bf16* __restrict__ Wih,   // [4H x IN] bf16 row-major
                     const __bf16* __restrict__ Whh,   // [4H x HD] bf16 row-major
                     const float* __restrict__ bih,
                     const float* __restrict__ bhh,
                     const __bf16* __restrict__ hprev, // [BT x HD]
                     __bf16* __restrict__ hcur,        // [BT x HD]
                     __bf16* __restrict__ hrelu,       // [BT x HD]
                     float* __restrict__ cst)          // [BT x HD]
{
    const int tid    = threadIdx.x;
    const int wave   = tid >> 5;
    const int lane   = tid & 31;
    const int lane16 = lane & 15;
    const int hi     = lane >> 4;
    const int wm     = wave >> 1;   // 0..3
    const int wn     = wave & 1;    // 0..1

    const int b0 = blockIdx.x * 64;        // batch tile base
    const int n0 = blockIdx.y * 64;        // H-column tile base
    const int rb = b0 + wm * 16;           // this wave's batch rows
    const int cb = n0 + wn * 32;           // this wave's 32 H-cols

    v8f acc[4][2];
#pragma unroll
    for (int g = 0; g < 4; ++g)
#pragma unroll
        for (int nt = 0; nt < 2; ++nt) acc[g][nt] = (v8f){};

    const int arow = rb + lane16;          // A-fragment: lane holds one batch row

    // ---- K chunks 0..3 : x_t (f32 in global, convert on the fly) ----
#pragma unroll
    for (int kc = 0; kc < 4; ++kc) {
        const float* xrow = x + ((size_t)arow * TT + t) * IN + kc * 32 + 16 * hi;
        AFrag a;
#pragma unroll
        for (int i = 0; i < 8; ++i) {
            a.v[i]     = (__bf16)xrow[i];
            a.v[8 + i] = (__bf16)xrow[8 + i];
        }
#pragma unroll
        for (int g = 0; g < 4; ++g) {
#pragma unroll
            for (int nt = 0; nt < 2; ++nt) {
                const int j = g * HD + cb + nt * 16 + lane16;       // W row (B column)
                const v16bf bm = *(const v16bf*)(Wih + (size_t)j * IN + kc * 32 + 16 * hi);
                acc[g][nt] = __builtin_amdgcn_wmma_f32_16x16x32_bf16(
                    false, a.v, false, bm, (short)0, acc[g][nt], false, false);
            }
        }
    }

    // ---- K chunks 0..15 : h_prev (bf16) ----
#pragma unroll 4
    for (int kc = 0; kc < 16; ++kc) {
        const __bf16* hrow = hprev + (size_t)arow * HD + kc * 32 + 16 * hi;
        AFrag a;
        a.h8[0] = *(const v8bf*)(hrow);
        a.h8[1] = *(const v8bf*)(hrow + 8);
#pragma unroll
        for (int g = 0; g < 4; ++g) {
#pragma unroll
            for (int nt = 0; nt < 2; ++nt) {
                const int j = g * HD + cb + nt * 16 + lane16;
                const v16bf bm = *(const v16bf*)(Whh + (size_t)j * HD + kc * 32 + 16 * hi);
                acc[g][nt] = __builtin_amdgcn_wmma_f32_16x16x32_bf16(
                    false, a.v, false, bm, (short)0, acc[g][nt], false, false);
            }
        }
    }

    // ---- epilogue: bias + activations + cell update ----
#pragma unroll
    for (int nt = 0; nt < 2; ++nt) {
        const int col = cb + nt * 16 + lane16;
        const float bI = bih[0 * HD + col] + bhh[0 * HD + col];
        const float bF = bih[1 * HD + col] + bhh[1 * HD + col];
        const float bG = bih[2 * HD + col] + bhh[2 * HD + col];
        const float bO = bih[3 * HD + col] + bhh[3 * HD + col];
#pragma unroll
        for (int e = 0; e < 8; ++e) {
            const int row = rb + e + 8 * hi;
            const float gi = sigmoidf_(acc[0][nt][e] + bI);
            const float gf = sigmoidf_(acc[1][nt][e] + bF);
            const float gg = htanh_(acc[2][nt][e] + bG);
            const float go = sigmoidf_(acc[3][nt][e] + bO);
            const size_t idx = (size_t)row * HD + col;
            const float cn = gf * cst[idx] + gi * gg;
            cst[idx] = cn;
            const float hn = go * htanh_(cn);
            hcur[idx]  = (__bf16)hn;
            hrelu[idx] = (__bf16)fmaxf(hn, 0.0f);
        }
    }
}

// ---------------------------------------------------------------------------
// Kernel 2: z = relu(h) @ fc1_w^T + fc1_b  -> relu -> dot fc_w + fc_b,
// accumulate MSE loss. One workgroup = 64 batch rows x all 256 fc1 outputs.
// ---------------------------------------------------------------------------
__global__ __launch_bounds__(256)
void lstm_fc_loss(const __bf16* __restrict__ hrelu,   // [BT x HD]
                  const __bf16* __restrict__ fc1b,    // [TT x HD] bf16
                  const float* __restrict__ fc1bias,  // [TT]
                  const float* __restrict__ fcw,      // [TT] (fc_w[0,:])
                  const float* __restrict__ fcb,      // [1]
                  const float* __restrict__ y,        // [TT x BT]
                  int t,
                  float* __restrict__ out,            // d_out[0..BT-1]
                  float* __restrict__ loss)           // d_out[BT]
{
    __shared__ float rowsum[64];

    const int tid    = threadIdx.x;
    const int wave   = tid >> 5;
    const int lane   = tid & 31;
    const int lane16 = lane & 15;
    const int hi     = lane >> 4;
    const int wm     = wave >> 1;   // 0..3 -> 16-row tiles
    const int wn     = wave & 1;    // 0..1 -> 128-col halves

    if (tid < 64) rowsum[tid] = 0.0f;
    __syncthreads();

    const int b0 = blockIdx.x * 64;
    const int rb = b0 + wm * 16;
    const int arow = rb + lane16;

    v8f acc[8];
#pragma unroll
    for (int nt = 0; nt < 8; ++nt) acc[nt] = (v8f){};

#pragma unroll 4
    for (int kc = 0; kc < 16; ++kc) {
        const __bf16* hrow = hrelu + (size_t)arow * HD + kc * 32 + 16 * hi;
        AFrag a;
        a.h8[0] = *(const v8bf*)(hrow);
        a.h8[1] = *(const v8bf*)(hrow + 8);
#pragma unroll
        for (int nt = 0; nt < 8; ++nt) {
            const int j = wn * 128 + nt * 16 + lane16;     // fc1 output index
            const v16bf bm = *(const v16bf*)(fc1b + (size_t)j * HD + kc * 32 + 16 * hi);
            acc[nt] = __builtin_amdgcn_wmma_f32_16x16x32_bf16(
                false, a.v, false, bm, (short)0, acc[nt], false, false);
        }
    }

    // epilogue: relu + weighted row sum
    float bias[8], wgt[8];
#pragma unroll
    for (int nt = 0; nt < 8; ++nt) {
        const int col = wn * 128 + nt * 16 + lane16;
        bias[nt] = fc1bias[col];
        wgt[nt]  = fcw[col];
    }
#pragma unroll
    for (int e = 0; e < 8; ++e) {
        float s = 0.0f;
#pragma unroll
        for (int nt = 0; nt < 8; ++nt) {
            float z = acc[nt][e] + bias[nt];
            z = fmaxf(z, 0.0f);
            s += z * wgt[nt];
        }
        atomicAdd(&rowsum[wm * 16 + e + 8 * hi], s);
    }
    __syncthreads();

    if (tid < 64) {
        const int b = b0 + tid;
        const float yp = rowsum[tid] + fcb[0];
        out[b] = yp;                                  // last step's write survives
        const float d = y[(size_t)t * BT + b] - yp;
        atomicAdd(loss, d * d * (1.0f / (float)BT));
    }
}

// ---------------------------------------------------------------------------
// host side
// ---------------------------------------------------------------------------
extern "C" void kernel_launch(void* const* d_in, const int* in_sizes, int n_in,
                              void* d_out, int out_size, void* d_ws, size_t ws_size,
                              hipStream_t stream) {
    (void)in_sizes; (void)n_in; (void)out_size; (void)ws_size;

    const float* x    = (const float*)d_in[0];   // [BT, TT, IN]
    const float* y    = (const float*)d_in[1];   // [TT, BT, 1]
    const float* Wih  = (const float*)d_in[2];   // [4H, IN]
    const float* Whh  = (const float*)d_in[3];   // [4H, HD]
    const float* bih  = (const float*)d_in[4];
    const float* bhh  = (const float*)d_in[5];
    const float* fc1w = (const float*)d_in[6];   // [TT, HD]
    const float* fc1b = (const float*)d_in[7];
    const float* fcw  = (const float*)d_in[8];   // [1, TT]
    const float* fcb  = (const float*)d_in[9];
    float* out = (float*)d_out;                  // [BT + 1]

    // workspace layout (byte offsets, 256-aligned sizes)
    char* ws = (char*)d_ws;
    const size_t nWih = (size_t)4 * HD * IN;     // 262144
    const size_t nWhh = (size_t)4 * HD * HD;     // 1048576
    const size_t nFc1 = (size_t)TT * HD;         // 131072
    const size_t nHC  = (size_t)BT * HD;         // 262144

    __bf16* wihB  = (__bf16*)(ws);
    __bf16* whhB  = (__bf16*)(ws + 524288);
    __bf16* fc1B  = (__bf16*)(ws + 2621440);
    __bf16* h0    = (__bf16*)(ws + 2883584);
    __bf16* h1    = (__bf16*)(ws + 3407872);
    __bf16* hrel  = (__bf16*)(ws + 3932160);
    float*  cst   = (float*)(ws + 4456448);
    // total ~5.5 MiB

    // one-time conversions / zeroing (re-done each call -> deterministic)
    cvt_f32_to_bf16<<<(int)((nWih + 255) / 256), 256, 0, stream>>>(Wih,  wihB, (int)nWih);
    cvt_f32_to_bf16<<<(int)((nWhh + 255) / 256), 256, 0, stream>>>(Whh,  whhB, (int)nWhh);
    cvt_f32_to_bf16<<<(int)((nFc1 + 255) / 256), 256, 0, stream>>>(fc1w, fc1B, (int)nFc1);
    zero_bf16_k<<<(int)((nHC + 255) / 256), 256, 0, stream>>>(h0,  (int)nHC);
    zero_f32_k <<<(int)((nHC + 255) / 256), 256, 0, stream>>>(cst, (int)nHC);
    zero_f32_k <<<1, 32, 0, stream>>>(out + BT, 1);   // loss accumulator

    dim3 gridG(BT / 64, HD / 64, 1);   // 8 x 8
    dim3 gridF(BT / 64, 1, 1);         // 8

    for (int t = 0; t < TT; ++t) {
        const __bf16* hp = (t & 1) ? h1 : h0;
        __bf16*       hc = (t & 1) ? h0 : h1;
        lstm_gates_cell<<<gridG, 256, 0, stream>>>(
            x, t, wihB, whhB, bih, bhh, hp, hc, hrel, cst);
        lstm_fc_loss<<<gridF, 256, 0, stream>>>(
            hrel, fc1B, fc1b, fcw, fcb, y, t, out, out + BT);
    }
}
